// StandardAttention_48979807044081
// MI455X (gfx1250) — matrix operand order repaired and checked by reading, not compiled, not run on hardware
//
#include <hip/hip_runtime.h>
#include <hip/hip_bf16.h>

#define SEQ      2048
#define HIDDEN   2048
#define KVH      512
#define NHEADS   32
#define NKVHEADS 8
#define HDIM     64
#define BATCH    2

typedef __attribute__((ext_vector_type(16))) __bf16       v16bf;
typedef __attribute__((ext_vector_type(8)))  float        v8f;
typedef __attribute__((ext_vector_type(4)))  unsigned int u32x4;

union BFrag {
  v16bf v;
  u32x4 u[2];
};

__device__ __forceinline__ unsigned short f2bf(float x) {
  unsigned int u = __builtin_bit_cast(unsigned int, x);
  u += 0x7FFFu + ((u >> 16) & 1u);   // round-to-nearest-even
  return (unsigned short)(u >> 16);
}

__device__ __forceinline__ v8f v8f_zero() {
  v8f z = {0.f, 0.f, 0.f, 0.f, 0.f, 0.f, 0.f, 0.f};
  return z;
}

__device__ __forceinline__ v8f wmma_bf16(v16bf a, v16bf b, v8f c) {
  // 8 args: (neg_a, A, neg_b, B, c_mod, C, reuse_a, reuse_b)
  return __builtin_amdgcn_wmma_f32_16x16x32_bf16(false, a, false, b, (short)0, c,
                                                 false, false);
}

// ---------------------------------------------------------------------------
// Elementwise f32 -> bf16 (vectorized x4)
// ---------------------------------------------------------------------------
__global__ __launch_bounds__(256) void f32_to_bf16_k(const float* __restrict__ in,
                                                     unsigned short* __restrict__ out,
                                                     int n4) {
  int i = blockIdx.x * blockDim.x + threadIdx.x;
  if (i >= n4) return;
  float4 v = ((const float4*)in)[i];
  ushort4 o;
  o.x = f2bf(v.x);
  o.y = f2bf(v.y);
  o.z = f2bf(v.z);
  o.w = f2bf(v.w);
  ((ushort4*)out)[i] = o;
}

// ---------------------------------------------------------------------------
// Tiled transpose f32[rows][cols] -> bf16[cols][rows], with fused scale
// (scale = 1/sqrt(HEAD_DIM) for Wq, folding the softmax scale into Q)
// ---------------------------------------------------------------------------
__global__ __launch_bounds__(256) void transpose_f32_to_bf16(
    const float* __restrict__ in, unsigned short* __restrict__ out,
    int rows, int cols, float scale) {
  __shared__ float tile[32][33];
  const int c0 = blockIdx.x * 32, r0 = blockIdx.y * 32;
  const int tx = threadIdx.x, ty = threadIdx.y;   // block (32,8)
  #pragma unroll
  for (int i = ty; i < 32; i += 8)
    tile[i][tx] = in[(size_t)(r0 + i) * cols + c0 + tx];
  __syncthreads();
  #pragma unroll
  for (int i = ty; i < 32; i += 8)
    out[(size_t)(c0 + i) * rows + r0 + tx] = f2bf(tile[tx][i] * scale);
}

// ---------------------------------------------------------------------------
// Tiled transpose bf16[rows][cols] -> bf16[cols][rows], batched over z
// ---------------------------------------------------------------------------
__global__ __launch_bounds__(256) void transpose_bf16_k(
    const unsigned short* __restrict__ in, unsigned short* __restrict__ out,
    int rows, int cols) {
  __shared__ unsigned short tile[32][33];
  const unsigned short* inb = in + (size_t)blockIdx.z * rows * cols;
  unsigned short* outb = out + (size_t)blockIdx.z * rows * cols;
  const int c0 = blockIdx.x * 32, r0 = blockIdx.y * 32;
  const int tx = threadIdx.x, ty = threadIdx.y;   // block (32,8)
  #pragma unroll
  for (int i = ty; i < 32; i += 8)
    tile[i][tx] = inb[(size_t)(r0 + i) * cols + c0 + tx];
  __syncthreads();
  #pragma unroll
  for (int i = ty; i < 32; i += 8)
    outb[(size_t)(c0 + i) * rows + r0 + tx] = tile[tx][i];
}

// ---------------------------------------------------------------------------
// GEMM: C[M,N] = A[M,K] * B[K,N], with B supplied transposed (Bt[N,K]) so
// both WMMA operands are contiguous 16B loads from global.
// Block = 256 thr = 8 waves stacked along M; wave computes a 32x128 tile
// (2 M-tiles x 8 N-tiles = 16 WMMA accumulators). All 8 waves share the
// same B fragments (WGP$-resident), so L2 traffic is A-dominated; the
// 8-wide N tiling amortizes each A fragment over 8 WMMAs.
// ---------------------------------------------------------------------------
template <bool F32OUT>
__global__ __launch_bounds__(256) void gemm_bf16_tn(
    const unsigned short* __restrict__ A,    // [M x K] bf16 row-major
    const unsigned short* __restrict__ Bt,   // [N x K] bf16 row-major (B^T)
    void* __restrict__ C,                    // [M x N]
    int M, int N, int K) {
  const int lane = threadIdx.x & 31;
  const int wid  = threadIdx.x >> 5;
  const int g = lane >> 4, ln = lane & 15;
  const int mbase = blockIdx.y * 256 + wid * 32;
  const int nbase = blockIdx.x * 128;

  v8f acc[2][8];
  #pragma unroll
  for (int i = 0; i < 2; ++i)
    #pragma unroll
    for (int j = 0; j < 8; ++j) acc[i][j] = v8f_zero();

  for (int k0 = 0; k0 < K; k0 += 32) {
    BFrag a[2];
    #pragma unroll
    for (int mt = 0; mt < 2; ++mt) {
      // A-frag: row = mbase+mt*16+ln; K = g*16+0..15
      const unsigned short* ap = A + (size_t)(mbase + mt * 16 + ln) * K + k0;
      a[mt].u[0] = *(const u32x4*)(ap + g * 16);
      a[mt].u[1] = *(const u32x4*)(ap + g * 16 + 8);
    }
    #pragma unroll
    for (int nt = 0; nt < 8; ++nt) {
      // B-frag: col = nbase+nt*16+ln; halves K = g*16+0..15 (row of Bt)
      const unsigned short* bp =
          Bt + (size_t)(nbase + nt * 16 + ln) * K + k0 + g * 16;
      BFrag b;
      b.u[0] = *(const u32x4*)(bp);
      b.u[1] = *(const u32x4*)(bp + 8);
      acc[0][nt] = wmma_bf16(a[0].v, b.v, acc[0][nt]);
      acc[1][nt] = wmma_bf16(a[1].v, b.v, acc[1][nt]);
    }
  }

  #pragma unroll
  for (int mt = 0; mt < 2; ++mt)
    #pragma unroll
    for (int nt = 0; nt < 8; ++nt)
      #pragma unroll
      for (int r = 0; r < 8; ++r) {
        // C/D layout: VGPR r, lane group g -> row r+8g, col = ln
        const int row = mbase + mt * 16 + r + 8 * g;
        const int col = nbase + nt * 16 + ln;
        const float v = acc[mt][nt][r];
        if (F32OUT)
          ((float*)C)[(size_t)row * N + col] = v;
        else
          ((unsigned short*)C)[(size_t)row * N + col] = f2bf(v);
      }
}

// ---------------------------------------------------------------------------
// Causal GQA flash attention. Grid (S/128, NHEADS, BATCH), 256 thr = 8 waves.
// Each wave owns one 16-row Q tile; iterates over 32-key tiles up to the
// causal diagonal with f32 online softmax. Q is pre-scaled by 1/sqrt(d)
// (folded into Wq), so scores come out of WMMA already scaled.
//   Q : [B, S, HIDDEN] bf16 (head = 64-col slice)  -> A-operand, direct loads
//   K : [B, S, KVH]    bf16                         -> B-operand of Q*K^T
//   Vt: [B, KVH, S]    bf16 (V transposed)          -> B-operand of P*V
//   O : [B, S, HIDDEN] bf16
// ---------------------------------------------------------------------------
__global__ __launch_bounds__(256) void flash_attn(
    const unsigned short* __restrict__ Q, const unsigned short* __restrict__ Kc,
    const unsigned short* __restrict__ Vt, unsigned short* __restrict__ O) {
  const int lane = threadIdx.x & 31;
  const int wid  = threadIdx.x >> 5;
  const int g = lane >> 4, ln = lane & 15;
  const int b = blockIdx.z;
  const int head = blockIdx.y;
  const int kvh = head >> 2;   // 32 q-heads over 8 kv-heads (n_rep = 4)
  const int qbase = blockIdx.x * 128 + wid * 16;

  __shared__ unsigned short pbuf[8][16][32];   // per-wave P staging (8 KB)

  // Q fragments (16 x 64, two 16x32 A-operands), kept in registers
  BFrag qa[2];
  {
    const unsigned short* qp =
        Q + (size_t)(b * SEQ + qbase + ln) * HIDDEN + head * HDIM;
    #pragma unroll
    for (int c = 0; c < 2; ++c) {
      qa[c].u[0] = *(const u32x4*)(qp + 32 * c + g * 16);
      qa[c].u[1] = *(const u32x4*)(qp + 32 * c + g * 16 + 8);
    }
  }
  const unsigned short* kbase = Kc + (size_t)b * SEQ * KVH + kvh * HDIM;
  const unsigned short* vbase = Vt + ((size_t)b * KVH + kvh * HDIM) * SEQ;

  float m[8], l[8];
  v8f acc[4];
  #pragma unroll
  for (int r = 0; r < 8; ++r) { m[r] = -1e30f; l[r] = 0.f; }
  #pragma unroll
  for (int j = 0; j < 4; ++j) acc[j] = v8f_zero();

  const int ktEnd = (qbase + 15) >> 5;   // inclusive causal tile bound
  for (int kt = 0; kt <= ktEnd; ++kt) {
    const int k0 = kt * 32;

    // ---- S = (Q * K^T): two 16x16 tiles, K-dim = 64 (two bf16 WMMAs each)
    v8f s[2];
    s[0] = v8f_zero();
    s[1] = v8f_zero();
    #pragma unroll
    for (int t = 0; t < 2; ++t) {
      #pragma unroll
      for (int c = 0; c < 2; ++c) {
        BFrag kb;
        const unsigned short* kp =
            kbase + (size_t)(k0 + 16 * t + ln) * KVH + 32 * c + g * 16;
        kb.u[0] = *(const u32x4*)(kp);
        kb.u[1] = *(const u32x4*)(kp + 8);
        s[t] = wmma_bf16(qa[c].v, kb.v, s[t]);
      }
    }

    // ---- causal mask (only the diagonal tile needs masking)
    const bool diag = (k0 + 31 > qbase);
    float rowm[8];
    #pragma unroll
    for (int r = 0; r < 8; ++r) {
      #pragma unroll
      for (int t = 0; t < 2; ++t) {
        float sv = s[t][r];
        if (diag && (k0 + 16 * t + ln > qbase + r + 8 * g)) sv = -1e30f;
        s[t][r] = sv;
      }
      rowm[r] = fmaxf(s[0][r], s[1][r]);
    }
    // row max across the 16 lanes of each half-wave (xor<16 stays in group)
    #pragma unroll
    for (int off = 1; off < 16; off <<= 1)
      #pragma unroll
      for (int r = 0; r < 8; ++r)
        rowm[r] = fmaxf(rowm[r], __shfl_xor(rowm[r], off, 32));

    float corr[8];
    #pragma unroll
    for (int r = 0; r < 8; ++r) {
      const float mn = fmaxf(m[r], rowm[r]);
      corr[r] = __expf(m[r] - mn);
      m[r] = mn;
    }

    // ---- P = exp(S - m): accumulate row sums, stage bf16 P in LDS
    float rs[8];
    #pragma unroll
    for (int r = 0; r < 8; ++r) {
      const float p0 = __expf(s[0][r] - m[r]);
      const float p1 = __expf(s[1][r] - m[r]);
      rs[r] = p0 + p1;
      pbuf[wid][r + 8 * g][ln]      = f2bf(p0);   // C-layout -> [row][col]
      pbuf[wid][r + 8 * g][16 + ln] = f2bf(p1);
    }
    #pragma unroll
    for (int off = 1; off < 16; off <<= 1)
      #pragma unroll
      for (int r = 0; r < 8; ++r)
        rs[r] += __shfl_xor(rs[r], off, 32);

    #pragma unroll
    for (int r = 0; r < 8; ++r) l[r] = l[r] * corr[r] + rs[r];
    #pragma unroll
    for (int j = 0; j < 4; ++j)
      #pragma unroll
      for (int r = 0; r < 8; ++r) acc[j][r] *= corr[r];

    // wave-local LDS RAW: P stores -> A-fragment loads
    asm volatile("s_wait_dscnt 0" ::: "memory");

    // ---- reload P in A-operand layout (16x32)
    BFrag pa;
    {
      const unsigned short* pp = &pbuf[wid][ln][0];
      pa.u[0] = *(const u32x4*)(pp + g * 16);
      pa.u[1] = *(const u32x4*)(pp + g * 16 + 8);
    }

    // ---- O += P * V : four 16x16 output tiles over d, K-dim = 32 keys
    #pragma unroll
    for (int j = 0; j < 4; ++j) {
      BFrag vb;
      const unsigned short* vp =
          vbase + (size_t)(16 * j + ln) * SEQ + k0 + g * 16;
      vb.u[0] = *(const u32x4*)(vp);
      vb.u[1] = *(const u32x4*)(vp + 8);
      acc[j] = wmma_bf16(pa.v, vb.v, acc[j]);
    }
  }

  // ---- epilogue: O = acc / l, store bf16 into [B, S, HIDDEN]
  unsigned short* op = O + (size_t)(b * SEQ + qbase) * HIDDEN + head * HDIM;
  #pragma unroll
  for (int j = 0; j < 4; ++j)
    #pragma unroll
    for (int r = 0; r < 8; ++r) {
      const float v = acc[j][r] / l[r];
      op[(size_t)(r + 8 * g) * HIDDEN + 16 * j + ln] = f2bf(v);
    }
}

// ---------------------------------------------------------------------------
// Host launcher
// ---------------------------------------------------------------------------
extern "C" void kernel_launch(void* const* d_in, const int* in_sizes, int n_in,
                              void* d_out, int out_size, void* d_ws,
                              size_t ws_size, hipStream_t stream) {
  (void)in_sizes; (void)n_in; (void)out_size; (void)ws_size;
  const float* X  = (const float*)d_in[0];
  const float* Wq = (const float*)d_in[1];
  const float* Wk = (const float*)d_in[2];
  const float* Wv = (const float*)d_in[3];
  const float* Wo = (const float*)d_in[4];
  // d_in[5] = layer_idx (unused)

  char* ws = (char*)d_ws;
  size_t off = 0;
  auto alloc = [&](size_t bytes) -> char* {
    char* p = ws + off;
    off += (bytes + 255) & ~(size_t)255;
    return p;
  };

  const int M = BATCH * SEQ;   // 4096
  unsigned short* Xb  = (unsigned short*)alloc((size_t)M * HIDDEN * 2);  // 16 MB
  unsigned short* Qb  = (unsigned short*)alloc((size_t)M * HIDDEN * 2);  // 16 MB
  unsigned short* Kb  = (unsigned short*)alloc((size_t)M * KVH * 2);     //  4 MB
  unsigned short* Vb  = (unsigned short*)alloc((size_t)M * KVH * 2);     //  4 MB
  unsigned short* Vtb = (unsigned short*)alloc((size_t)M * KVH * 2);     //  4 MB
  unsigned short* Ob  = (unsigned short*)alloc((size_t)M * HIDDEN * 2);  // 16 MB
  unsigned short* Wqt = (unsigned short*)alloc((size_t)HIDDEN * HIDDEN * 2);
  unsigned short* Wkt = (unsigned short*)alloc((size_t)HIDDEN * KVH * 2);
  unsigned short* Wvt = (unsigned short*)alloc((size_t)HIDDEN * KVH * 2);
  unsigned short* Wot = (unsigned short*)alloc((size_t)HIDDEN * HIDDEN * 2);

  // 1) X -> bf16
  {
    const int n4 = M * HIDDEN / 4;
    f32_to_bf16_k<<<(n4 + 255) / 256, 256, 0, stream>>>(X, Xb, n4);
  }
  // 2) weight transposes (f32 [K][N] -> bf16 [N][K]); Wq carries 1/sqrt(d)
  {
    dim3 tb(32, 8);
    const float qscale = 0.125f;   // 1/sqrt(HEAD_DIM=64)
    transpose_f32_to_bf16<<<dim3(HIDDEN / 32, HIDDEN / 32), tb, 0, stream>>>(
        Wq, Wqt, HIDDEN, HIDDEN, qscale);
    transpose_f32_to_bf16<<<dim3(KVH / 32, HIDDEN / 32), tb, 0, stream>>>(
        Wk, Wkt, HIDDEN, KVH, 1.0f);
    transpose_f32_to_bf16<<<dim3(KVH / 32, HIDDEN / 32), tb, 0, stream>>>(
        Wv, Wvt, HIDDEN, KVH, 1.0f);
    transpose_f32_to_bf16<<<dim3(HIDDEN / 32, HIDDEN / 32), tb, 0, stream>>>(
        Wo, Wot, HIDDEN, HIDDEN, 1.0f);
  }
  // 3) projections (WMMA GEMMs), block tile 256(M) x 128(N)
  gemm_bf16_tn<false><<<dim3(HIDDEN / 128, M / 256), 256, 0, stream>>>(
      Xb, Wqt, Qb, M, HIDDEN, HIDDEN);
  gemm_bf16_tn<false><<<dim3(KVH / 128, M / 256), 256, 0, stream>>>(
      Xb, Wkt, Kb, M, KVH, HIDDEN);
  gemm_bf16_tn<false><<<dim3(KVH / 128, M / 256), 256, 0, stream>>>(
      Xb, Wvt, Vb, M, KVH, HIDDEN);
  // 4) V -> V^T per batch ([S][KVH] -> [KVH][S])
  {
    dim3 tb(32, 8);
    transpose_bf16_k<<<dim3(KVH / 32, SEQ / 32, BATCH), tb, 0, stream>>>(
        Vb, Vtb, SEQ, KVH);
  }
  // 5) causal GQA flash attention (WMMA)
  flash_attn<<<dim3(SEQ / 128, NHEADS, BATCH), 256, 0, stream>>>(Qb, Kb, Vtb,
                                                                 Ob);
  // 6) output projection -> f32 result
  gemm_bf16_tn<true><<<dim3(HIDDEN / 128, M / 256), 256, 0, stream>>>(
      Ob, Wot, (void*)d_out, M, HIDDEN, HIDDEN);
}
